// ShiftedWindowAttention_69818988363869
// MI455X (gfx1250) — matrix operand order repaired and checked by reading, not compiled, not run on hardware
//
#include <hip/hip_runtime.h>

typedef _Float16 f16;
typedef __attribute__((ext_vector_type(16))) _Float16 v16h;
typedef __attribute__((ext_vector_type(8)))  float    v8f;

#define WSZ   8
#define SHIFT 4
#define NH    8
#define C_    256
#define H_    128
#define W_    128
#define HD    32

// Padded LDS leading dims (in elements) chosen so row strides are not
// multiples of 64 dwords -> avoid LDS bank conflicts on frag loads.
#define QKV_LD 776
#define XW_LD  260
#define O_LD   264
#define P_LD   68
#define STG_LD 260

// LDS byte offsets (phase-aliased regions)
#define OFF_QKV 0                      // 64*776*2  = 99328   (phase1..2)
#define OFF_XW  99328                  // 64*260*2  = 33280   (phase0..1)
#define OFF_O   99328                  // 64*264*2  = 33792   (phase2..3), aliases XW
#define OFF_P   133120                 // 8*16*68*2 = 17408   (phase2)
#define OFF_STG 0                      // 64*260*4  = 66560   (phase3..4), aliases QKV
#define SMEM_BYTES 150528

// ---------------- WMMA fragment helpers (wave32, CDNA5 layouts) -------------

// A-matrix 16x32 f16: lanes 0-15 rows M=0..15; lanes 0-15 hold K=0..15,
// lanes 16-31 hold K=16..31; VGPR j holds K = 2j, 2j+1 within the half.
__device__ __forceinline__ v16h load_a(const f16* __restrict__ p, int ld,
                                       int row0, int k0, int lane) {
  int r = lane & 15, hi = lane >> 4;
  const f16* base = p + (row0 + r) * ld + k0 + hi * 16;
  v16h f;
#pragma unroll
  for (int j = 0; j < 16; ++j) f[j] = base[j];
  return f;
}

// B-matrix 32x16 f16 from B^T stored row-major [n][k] (e.g. w_qkv/w_proj, or
// K-matrix as rows of tokens): lane = column n; lanes 0-15 hold K=0..15,
// lanes 16-31 hold K=16..31; VGPR j holds K = 2j, 2j+1 within the half.
__device__ __forceinline__ v16h load_b_nk(const f16* __restrict__ p, int ld,
                                          int n0, int k0, int lane) {
  int c = lane & 15, hi = lane >> 4;
  const f16* base = p + (n0 + c) * ld + k0 + hi * 16;
  v16h f;
#pragma unroll
  for (int j = 0; j < 16; ++j) f[j] = base[j];
  return f;
}

// B-matrix 32x16 f16 from B stored row-major [k][n] (e.g. V: rows=tokens).
__device__ __forceinline__ v16h load_b_kn(const f16* __restrict__ p, int ld,
                                          int n0, int k0, int lane) {
  int c = lane & 15, hi = lane >> 4;
  const f16* base = p + (k0 + hi * 16) * ld + n0 + c;
  v16h f;
#pragma unroll
  for (int j = 0; j < 16; ++j) f[j] = base[j * ld];
  return f;
}

__device__ __forceinline__ v8f wmma16(v16h a, v16h b, v8f c) {
  return __builtin_amdgcn_wmma_f32_16x16x32_f16(false, a, false, b,
                                                (short)0, c, false, false);
}

// 16-lane (half-wave) reductions; C/D frag rows live in 16-lane groups.
__device__ __forceinline__ float rmax16(float v) {
  v = fmaxf(v, __shfl_xor(v, 1, 32));
  v = fmaxf(v, __shfl_xor(v, 2, 32));
  v = fmaxf(v, __shfl_xor(v, 4, 32));
  v = fmaxf(v, __shfl_xor(v, 8, 32));
  return v;
}
__device__ __forceinline__ float rsum16(float v) {
  v += __shfl_xor(v, 1, 32);
  v += __shfl_xor(v, 2, 32);
  v += __shfl_xor(v, 4, 32);
  v += __shfl_xor(v, 8, 32);
  return v;
}

// Swin shift-mask region id for an (unshifted) grid position.
__device__ __forceinline__ int reg3(int v) {
  return v < (H_ - WSZ) ? 0 : (v < (H_ - SHIFT) ? 1 : 2);
}
__device__ __forceinline__ int tok_id(int t, int wh, int wwn) {
  return reg3(wh * WSZ + (t >> 3)) * 3 + reg3(wwn * WSZ + (t & 7));
}

// ---------------------------- kernels ---------------------------------------

__global__ void convert_weights(const float* __restrict__ wq,
                                const float* __restrict__ wp,
                                f16* __restrict__ wq16, f16* __restrict__ wp16) {
  int i = blockIdx.x * blockDim.x + threadIdx.x;
  if (i < 3 * C_ * C_) wq16[i] = (f16)wq[i];
  if (i < C_ * C_)     wp16[i] = (f16)wp[i];
}

__global__ void __launch_bounds__(256)
swin_fused(const float* __restrict__ x, const f16* __restrict__ wqkv,
           const float* __restrict__ bqkv, const f16* __restrict__ wproj,
           const float* __restrict__ bproj, float* __restrict__ out) {
  extern __shared__ char smem[];
  f16*   qkv = (f16*)(smem + OFF_QKV);   // 64 x 768 (+pad), f16
  f16*   xw  = (f16*)(smem + OFF_XW);    // 64 x 256 (+pad), f16
  f16*   olb = (f16*)(smem + OFF_O);     // 64 x 256 (+pad), f16
  float* stg = (float*)(smem + OFF_STG); // 64 x 256 (+pad), f32

  const int tid  = threadIdx.x;
  const int lane = tid & 31;
  const int wv   = tid >> 5;
  const int r16  = lane & 15;
  const int hi   = lane >> 4;

  const int b    = blockIdx.x >> 8;   // 256 windows per image
  const int wIdx = blockIdx.x & 255;
  const int wh   = wIdx >> 4;
  const int wwn  = wIdx & 15;

  const v8f vzero = {0.f, 0.f, 0.f, 0.f, 0.f, 0.f, 0.f, 0.f};

  // ---- Phase 0: shifted-window gather (B,C,H,W) fp32 -> LDS f16 ----
  {
    const float* xb = x + (size_t)b * C_ * H_ * W_;
#pragma unroll 4
    for (int it = 0; it < 64; ++it) {
      int idx = it * 256 + tid;
      int c = idx >> 6;
      int t = idx & 63;
      int hh = (wh * WSZ + (t >> 3) + SHIFT) & (H_ - 1);
      int ww = (wwn * WSZ + (t & 7) + SHIFT) & (W_ - 1);
      xw[t * XW_LD + c] = (f16)xb[(c * H_ + hh) * W_ + ww];
    }
  }
  __syncthreads();

  // ---- Phase 1: QKV GEMM (64x256)@(256x768) + bias -> qkv LDS (f16) ----
  for (int nt = wv * 6; nt < wv * 6 + 6; ++nt) {
    v8f acc[4] = {vzero, vzero, vzero, vzero};
    for (int ks = 0; ks < 8; ++ks) {
      v16h bf = load_b_nk(wqkv, C_, nt * 16, ks * 32, lane);
#pragma unroll
      for (int mt = 0; mt < 4; ++mt) {
        v16h af = load_a(xw, XW_LD, mt * 16, ks * 32, lane);
        acc[mt] = wmma16(af, bf, acc[mt]);
      }
    }
    float bias = bqkv[nt * 16 + r16];
#pragma unroll
    for (int mt = 0; mt < 4; ++mt)
#pragma unroll
      for (int j = 0; j < 8; ++j)
        qkv[(mt * 16 + hi * 8 + j) * QKV_LD + nt * 16 + r16] =
            (f16)(acc[mt][j] + bias);
  }
  __syncthreads();

  // ---- Phase 2: per-head attention (wave wv owns head wv) ----
  {
    const int h = wv;
    f16* pbase = (f16*)(smem + OFF_P) + wv * 16 * P_LD;  // 16x64 P scratch
    const f16* qh = qkv + h * HD;
    const f16* kh = qkv + C_ + h * HD;
    const f16* vh = qkv + 2 * C_ + h * HD;
    const float scale = 0.17677669529663687f;  // 1/sqrt(32)

    int idC[4];
#pragma unroll
    for (int nt = 0; nt < 4; ++nt) idC[nt] = tok_id(nt * 16 + r16, wh, wwn);

    for (int mt = 0; mt < 4; ++mt) {
      // S = Q Kt for one 16-row block (K-dim = hd = 32, single WMMA step)
      v16h aq = load_a(qh, QKV_LD, mt * 16, 0, lane);
      v8f s[4];
#pragma unroll
      for (int nt = 0; nt < 4; ++nt) {
        v16h bk = load_b_nk(kh, QKV_LD, nt * 16, 0, lane);
        s[nt] = wmma16(aq, bk, vzero);
      }
      // masked softmax over 64 columns (4 tiles x 16 lanes) per row
#pragma unroll
      for (int j = 0; j < 8; ++j) {
        int idR = tok_id(mt * 16 + hi * 8 + j, wh, wwn);
        float v0 = s[0][j] * scale + (idR == idC[0] ? 0.f : -100.f);
        float v1 = s[1][j] * scale + (idR == idC[1] ? 0.f : -100.f);
        float v2 = s[2][j] * scale + (idR == idC[2] ? 0.f : -100.f);
        float v3 = s[3][j] * scale + (idR == idC[3] ? 0.f : -100.f);
        float m = rmax16(fmaxf(fmaxf(v0, v1), fmaxf(v2, v3)));
        float e0 = __expf(v0 - m), e1 = __expf(v1 - m);
        float e2 = __expf(v2 - m), e3 = __expf(v3 - m);
        float inv = 1.0f / rsum16(e0 + e1 + e2 + e3);
        int rl = hi * 8 + j;
        pbase[rl * P_LD + 0  + r16] = (f16)(e0 * inv);
        pbase[rl * P_LD + 16 + r16] = (f16)(e1 * inv);
        pbase[rl * P_LD + 32 + r16] = (f16)(e2 * inv);
        pbase[rl * P_LD + 48 + r16] = (f16)(e3 * inv);
      }
      // O = P V  (K-dim = 64 tokens -> 2 WMMA steps; N = hd = 32 -> 2 tiles)
      v16h ap0 = load_a(pbase, P_LD, 0, 0, lane);
      v16h ap1 = load_a(pbase, P_LD, 0, 32, lane);
#pragma unroll
      for (int nt2 = 0; nt2 < 2; ++nt2) {
        v8f ao = wmma16(ap0, load_b_kn(vh, QKV_LD, nt2 * 16, 0, lane), vzero);
        ao     = wmma16(ap1, load_b_kn(vh, QKV_LD, nt2 * 16, 32, lane), ao);
#pragma unroll
        for (int j = 0; j < 8; ++j)
          olb[(mt * 16 + hi * 8 + j) * O_LD + h * HD + nt2 * 16 + r16] =
              (f16)ao[j];
      }
    }
  }
  __syncthreads();

  // ---- Phase 3: proj GEMM (64x256)@(256x256) + bias -> f32 staging ----
  for (int q = 0; q < 2; ++q) {
    int nt = wv * 2 + q;
    v8f acc[4] = {vzero, vzero, vzero, vzero};
    for (int ks = 0; ks < 8; ++ks) {
      v16h bf = load_b_nk(wproj, C_, nt * 16, ks * 32, lane);
#pragma unroll
      for (int mt = 0; mt < 4; ++mt) {
        v16h af = load_a(olb, O_LD, mt * 16, ks * 32, lane);
        acc[mt] = wmma16(af, bf, acc[mt]);
      }
    }
    float bias = bproj[nt * 16 + r16];
#pragma unroll
    for (int mt = 0; mt < 4; ++mt)
#pragma unroll
      for (int j = 0; j < 8; ++j)
        stg[(mt * 16 + hi * 8 + j) * STG_LD + nt * 16 + r16] =
            acc[mt][j] + bias;
  }
  __syncthreads();

  // ---- Phase 4: window-reverse + roll-back scatter to (B,C,H,W) fp32 ----
  {
    float* ob = out + (size_t)b * C_ * H_ * W_;
#pragma unroll 4
    for (int it = 0; it < 64; ++it) {
      int idx = it * 256 + tid;
      int c = idx >> 6;
      int t = idx & 63;
      int hh = (wh * WSZ + (t >> 3) + SHIFT) & (H_ - 1);
      int ww = (wwn * WSZ + (t & 7) + SHIFT) & (W_ - 1);
      ob[(c * H_ + hh) * W_ + ww] = stg[t * STG_LD + c];
    }
  }
}

// ---------------------------- launch -----------------------------------------

extern "C" void kernel_launch(void* const* d_in, const int* in_sizes, int n_in,
                              void* d_out, int out_size, void* d_ws,
                              size_t ws_size, hipStream_t stream) {
  (void)in_sizes; (void)n_in; (void)out_size; (void)ws_size;
  const float* x     = (const float*)d_in[0];
  const float* wqkv  = (const float*)d_in[1];
  const float* bqkv  = (const float*)d_in[2];
  const float* wproj = (const float*)d_in[3];
  const float* bproj = (const float*)d_in[4];
  float* out = (float*)d_out;

  f16* wq16 = (f16*)d_ws;                 // 196608 halfs
  f16* wp16 = wq16 + 3 * C_ * C_;         // 65536 halfs

  convert_weights<<<1024, 256, 0, stream>>>(wqkv, wproj, wq16, wp16);

  (void)hipFuncSetAttribute((const void*)swin_fused,
                            hipFuncAttributeMaxDynamicSharedMemorySize,
                            SMEM_BYTES);
  swin_fused<<<8 * 256, 256, SMEM_BYTES, stream>>>(x, wq16, bqkv, wp16, bproj,
                                                   out);
}
